// MambaEncoder_86466281603118
// MI455X (gfx1250) — compile-verified, matrix operand order checked by reading, not run on hardware
//
#include <hip/hip_runtime.h>
#include <hip/hip_bf16.h>
#include <math.h>

// ---------------------------------------------------------------------------
// MambaEncoder for MI455X (gfx1250, wave32, WMMA).
// GEMMs: v_wmma_f32_16x16x32_bf16, double-buffered LDS, async global->LDS DMA
// (ASYNCcnt) when the toolchain exposes the gfx1250 async builtins.
// ---------------------------------------------------------------------------

typedef __attribute__((ext_vector_type(16))) __bf16 v16bf;
typedef __attribute__((ext_vector_type(8)))  float  v8f;

#define BATCH 4
#define SEQL  1024
#define DMOD  1024
#define DIN   2048
#define NST   16
#define RRANK 64
#define HMLP  8192
#define ROWS  (BATCH * SEQL)   // 4096

#ifndef __has_builtin
#define __has_builtin(x) 0
#endif

#if __has_builtin(__builtin_amdgcn_global_load_async_to_lds_b128)
#define USE_ASYNC_STAGE 1
#else
#define USE_ASYNC_STAGE 0
#endif

#if USE_ASYNC_STAGE
// Builtin signature (probe-confirmed): (AS1 int-vec16B*, AS3 int-vec16B*, imm, imm)
typedef int v4i_b128 __attribute__((__vector_size__(16)));
typedef __attribute__((address_space(1))) v4i_b128 as1_v4i;
typedef __attribute__((address_space(3))) v4i_b128 as3_v4i;

static __device__ __forceinline__ void async_cp_b128(__bf16* lds_dst, const __bf16* g_src)
{
    __builtin_amdgcn_global_load_async_to_lds_b128(
        (as1_v4i*)g_src, (as3_v4i*)lds_dst, 0, 0);
}

static __device__ __forceinline__ void wait_async0()
{
#if __has_builtin(__builtin_amdgcn_s_wait_asynccnt)
    __builtin_amdgcn_s_wait_asynccnt(0);
#else
    asm volatile("s_wait_asynccnt 0" ::: "memory");
#endif
}
#endif

// ---------------------------------------------------------------------------
// Generic bf16 WMMA GEMM:  C[M,N] = epi(A[M,K] * W[N,K]^T + bias) + residual
// Block: 256 threads (8 waves), tile 128x128, K-step 32, double-buffered LDS.
// Wave grid 2(M) x 4(N); each wave: 4x2 fragments of 16x16 -> 8 WMMA / k-step.
// LDS held in swizzled per-lane fragment layout: every 16B chunk of a tile row
// lands contiguously in one lane's fragment -> 2x ds_load_b128 per fragment,
// and each global 16B chunk maps to one 16B LDS chunk -> async-DMA friendly.
// Partial-N tiles: W row index is clamped (not zero-filled) -- column n of B
// only affects column n of D, and the epilogue masks stores with gn >= N, so
// clamped-duplicate rows are harmless and the hot loop stays branch-free.
// remap: A row index = m + (m/remapDiv)*remapMul + remapAdd  (for padded conv)
// epi: 0=none 1=softplus 2=gelu(exact/erf)
// ---------------------------------------------------------------------------
__global__ __launch_bounds__(256) void gemm_wmma_bf16(
    const __bf16* __restrict__ A, const __bf16* __restrict__ W,
    float* Cf, __bf16* Cbf,
    const float* __restrict__ bias, const float* __restrict__ residual,
    int M, int N, int K, int lda, int ldb, int ldc,
    int remapDiv, int remapMul, int remapAdd,
    int epi, int accumulate)
{
    __shared__ __align__(16) __bf16 lsA[2][8 * 32 * 16];  // 8KB per buffer
    __shared__ __align__(16) __bf16 lsB[2][8 * 32 * 16];

    const int tid    = threadIdx.x;
    const int lane   = tid & 31;
    const int wave   = tid >> 5;
    const int waveMi = wave >> 2;   // 0..1  (64 rows each)
    const int waveNi = wave & 3;    // 0..3  (32 cols each)
    const int mBase  = blockIdx.y * 128;
    const int nBase  = blockIdx.x * 128;

    // Per-thread staging geometry: 128 rows x 4 chunks(8 halves) = 512 slots,
    // 2 per thread. Chunk cc of row r goes to frag-group r/16, lane
    // (r%16)+16*(cc&1), byte offset 16*(cc>>1): matches the 16-bit A/B WMMA
    // VGPR layout so fragment loads are 2 contiguous b128s.
    int st_dst[2];
    size_t st_aoff[2], st_boff[2];
    #pragma unroll
    for (int s = 0; s < 2; ++s) {
        int chunk = tid * 2 + s;
        int row   = chunk >> 2;
        int cc    = chunk & 3;
        int laneD = (row & 15) + ((cc & 1) << 4);
        st_dst[s] = ((row >> 4) * 32 + laneD) * 16 + ((cc >> 1) << 3);

        int gm   = mBase + row;
        int arow = gm + (gm / remapDiv) * remapMul + remapAdd;
        st_aoff[s] = (size_t)arow * lda + cc * 8;

        int gn = nBase + row;
        if (gn > N - 1) gn = N - 1;           // clamp; see header comment
        st_boff[s] = (size_t)gn * ldb + cc * 8;
    }

    v8f c[4][2] = {};
    const int nk = K >> 5;

#if USE_ASYNC_STAGE
    // ---- issue tile 0 ----
    #pragma unroll
    for (int s = 0; s < 2; ++s) {
        async_cp_b128(&lsA[0][st_dst[s]], A + st_aoff[s]);
        async_cp_b128(&lsB[0][st_dst[s]], W + st_boff[s]);
    }

    for (int i = 0; i < nk; ++i) {
        const int cur = i & 1;
        wait_async0();        // own-wave DMA for buffer `cur` complete
        __syncthreads();      // everyone's DMA for `cur` visible; all reads of
                              // buffer cur^1 from iter i-1 are done.
        if (i + 1 < nk) {
            const int k0n = (i + 1) << 5;
            #pragma unroll
            for (int s = 0; s < 2; ++s) {
                async_cp_b128(&lsA[cur ^ 1][st_dst[s]], A + st_aoff[s] + k0n);
                async_cp_b128(&lsB[cur ^ 1][st_dst[s]], W + st_boff[s] + k0n);
            }
        }

        union FragBF { uint4 q[2]; v16bf v; };
        FragBF a[4], b[2];
        #pragma unroll
        for (int fn = 0; fn < 2; ++fn) {
            const __bf16* p = &lsB[cur][((waveNi * 2 + fn) * 32 + lane) * 16];
            b[fn].q[0] = *(const uint4*)p;
            b[fn].q[1] = *(const uint4*)(p + 8);
        }
        #pragma unroll
        for (int fm = 0; fm < 4; ++fm) {
            const __bf16* p = &lsA[cur][((waveMi * 4 + fm) * 32 + lane) * 16];
            a[fm].q[0] = *(const uint4*)p;
            a[fm].q[1] = *(const uint4*)(p + 8);
        }
        #pragma unroll
        for (int fm = 0; fm < 4; ++fm)
            #pragma unroll
            for (int fn = 0; fn < 2; ++fn)
                c[fm][fn] = __builtin_amdgcn_wmma_f32_16x16x32_bf16(
                    false, a[fm].v, false, b[fn].v, (short)0, c[fm][fn],
                    false, false);
    }
#else
    // ---- fallback: synchronous staging through VGPRs, single buffer ----
    for (int i = 0; i < nk; ++i) {
        const int k0 = i << 5;
        #pragma unroll
        for (int s = 0; s < 2; ++s) {
            uint4 va = *(const uint4*)(A + st_aoff[s] + k0);
            *(uint4*)&lsA[0][st_dst[s]] = va;
            uint4 vb = *(const uint4*)(W + st_boff[s] + k0);
            *(uint4*)&lsB[0][st_dst[s]] = vb;
        }
        __syncthreads();

        union FragBF { uint4 q[2]; v16bf v; };
        FragBF a[4], b[2];
        #pragma unroll
        for (int fn = 0; fn < 2; ++fn) {
            const __bf16* p = &lsB[0][((waveNi * 2 + fn) * 32 + lane) * 16];
            b[fn].q[0] = *(const uint4*)p;
            b[fn].q[1] = *(const uint4*)(p + 8);
        }
        #pragma unroll
        for (int fm = 0; fm < 4; ++fm) {
            const __bf16* p = &lsA[0][((waveMi * 4 + fm) * 32 + lane) * 16];
            a[fm].q[0] = *(const uint4*)p;
            a[fm].q[1] = *(const uint4*)(p + 8);
        }
        #pragma unroll
        for (int fm = 0; fm < 4; ++fm)
            #pragma unroll
            for (int fn = 0; fn < 2; ++fn)
                c[fm][fn] = __builtin_amdgcn_wmma_f32_16x16x32_bf16(
                    false, a[fm].v, false, b[fn].v, (short)0, c[fm][fn],
                    false, false);
        __syncthreads();
    }
#endif

    // ---- epilogue ----
    const int hi = lane >> 4;
    const int ln = lane & 15;
    #pragma unroll
    for (int fm = 0; fm < 4; ++fm) {
        #pragma unroll
        for (int fn = 0; fn < 2; ++fn) {
            int gn = nBase + waveNi * 32 + fn * 16 + ln;
            if (gn >= N) continue;
            #pragma unroll
            for (int i = 0; i < 8; ++i) {
                int gm   = mBase + waveMi * 64 + fm * 16 + hi * 8 + i;
                size_t o = (size_t)gm * ldc + gn;
                float v  = c[fm][fn][i];
                if (accumulate) v += Cf[o];
                if (bias)       v += bias[gn];
                if (epi == 1)      v = (v > 20.f) ? v : log1pf(__expf(v));
                else if (epi == 2) v = 0.5f * v * (1.f + erff(v * 0.70710678118f));
                if (residual)   v += residual[o];
                if (Cf)  Cf[o]  = v;
                if (Cbf) Cbf[o] = (__bf16)v;
            }
        }
    }
}

// ---------------------------------------------------------------------------
// LayerNorm over last dim (D=1024), one block per row, bf16 output.
// ---------------------------------------------------------------------------
__global__ __launch_bounds__(256) void layernorm_bf(
    const float* __restrict__ x, const float* __restrict__ w,
    const float* __restrict__ b, __bf16* __restrict__ out)
{
    const int row = blockIdx.x;
    const int tid = threadIdx.x;
    const float* xr = x + (size_t)row * DMOD;
    float4 v = *(const float4*)&xr[tid * 4];
    float s  = v.x + v.y + v.z + v.w;
    float ss = v.x * v.x + v.y * v.y + v.z * v.z + v.w * v.w;

    __shared__ float r1[256], r2[256];
    r1[tid] = s; r2[tid] = ss;
    __syncthreads();
    for (int off = 128; off > 0; off >>= 1) {
        if (tid < off) { r1[tid] += r1[tid + off]; r2[tid] += r2[tid + off]; }
        __syncthreads();
    }
    float mean = r1[0] * (1.f / DMOD);
    float var  = r2[0] * (1.f / DMOD) - mean * mean;
    float rstd = rsqrtf(var + 1e-6f);

    __bf16* orow = out + (size_t)row * DMOD;
    #pragma unroll
    for (int k = 0; k < 4; ++k) {
        int cidx = tid * 4 + k;
        float xv = (k == 0) ? v.x : (k == 1) ? v.y : (k == 2) ? v.z : v.w;
        orow[cidx] = (__bf16)((xv - mean) * rstd * w[cidx] + b[cidx]);
    }
}

// ---------------------------------------------------------------------------
// Causal depthwise conv (k=4) + SiLU over the xx half of xz. Output bf16.
// ---------------------------------------------------------------------------
__global__ __launch_bounds__(256) void dwconv_silu(
    const __bf16* __restrict__ xz, const float* __restrict__ conv_w,
    const float* __restrict__ conv_b, __bf16* __restrict__ xc)
{
    size_t idx = (size_t)blockIdx.x * 256 + threadIdx.x;   // over ROWS*DIN
    int d = idx % DIN;
    size_t r = idx / DIN;
    int l = (int)(r % SEQL);
    int bch = (int)(r / SEQL);
    float acc = conv_b[d];
    #pragma unroll
    for (int j = 0; j < 4; ++j) {
        int ls = l - 3 + j;
        if (ls >= 0)
            acc += conv_w[d * 4 + j] *
                   (float)xz[((size_t)(bch * SEQL + ls)) * (2 * DIN) + d];
    }
    float sv = acc / (1.f + __expf(-acc));                 // silu
    xc[idx] = (__bf16)sv;
}

// ---------------------------------------------------------------------------
// Selective scan: one thread per (batch, channel). h[16] in registers; B/C
// rows broadcast via LDS (double-buffered, one barrier per step). Fuses the
// +D*x skip and the silu(z) gate; emits bf16 for the out_proj GEMM.
// ---------------------------------------------------------------------------
__global__ __launch_bounds__(256) void selective_scan(
    const float* __restrict__ dt, const float* __restrict__ xdbl,
    const __bf16* __restrict__ xc, const __bf16* __restrict__ xz,
    const float* __restrict__ A_log, const float* __restrict__ Dp,
    __bf16* __restrict__ ybf)
{
    const int bch = blockIdx.y;
    const int d   = blockIdx.x * 256 + threadIdx.x;

    float Arow[NST];
    #pragma unroll
    for (int n = 0; n < NST; ++n) Arow[n] = -__expf(A_log[d * NST + n]);
    float h[NST];
    #pragma unroll
    for (int n = 0; n < NST; ++n) h[n] = 0.f;
    const float Dd = Dp[d];

    __shared__ float BC[2][32];   // [buf][0..15]=B, [16..31]=C
    for (int l = 0; l < SEQL; ++l) {
        int r   = bch * SEQL + l;
        int buf = l & 1;
        if (threadIdx.x < 32)
            BC[buf][threadIdx.x] = xdbl[(size_t)r * 96 + RRANK + threadIdx.x];
        __syncthreads();

        float dtv = dt[(size_t)r * DIN + d];
        float xv  = (float)xc[(size_t)r * DIN + d];
        float zv  = (float)xz[(size_t)r * (2 * DIN) + DIN + d];
        float dtx = dtv * xv;
        float y = 0.f;
        #pragma unroll
        for (int n = 0; n < NST; ++n) {
            float dA = __expf(dtv * Arow[n]);
            h[n] = dA * h[n] + dtx * BC[buf][n];
            y += h[n] * BC[buf][16 + n];
        }
        y = (y + Dd * xv) * (zv / (1.f + __expf(-zv)));
        ybf[(size_t)r * DIN + d] = (__bf16)y;
    }
}

// ---------------------------------------------------------------------------
// Small helpers: f32 -> bf16 converts, conv-weight tap split, padded x copy.
// ---------------------------------------------------------------------------
__global__ __launch_bounds__(256) void cvt_f32_bf16(
    const float* __restrict__ in, __bf16* __restrict__ out, int n)
{
    int i = blockIdx.x * 256 + threadIdx.x;
    if (i < n) out[i] = (__bf16)in[i];
}

// convL_w (D,D,5) -> 5 contiguous (D,D) bf16 tap matrices: out[t][n][k]
__global__ __launch_bounds__(256) void cvt_convL(
    const float* __restrict__ in, __bf16* __restrict__ out)
{
    int i = blockIdx.x * 256 + threadIdx.x;       // 5*D*D total
    if (i < 5 * DMOD * DMOD) {
        int t  = i / (DMOD * DMOD);
        int nk = i % (DMOD * DMOD);
        out[i] = (__bf16)in[(size_t)nk * 5 + t];
    }
}

// x (B,L,D) f32 -> zero-padded (B, L+4, D) bf16 (2 zero rows each side)
__global__ __launch_bounds__(256) void pad_x_bf(
    const float* __restrict__ x, __bf16* __restrict__ xpad)
{
    size_t idx = (size_t)blockIdx.x * 256 + threadIdx.x;  // B*(L+4)*D
    int dcol = idx % DMOD;
    size_t rr = idx / DMOD;
    int lp = (int)(rr % (SEQL + 4));
    int bch = (int)(rr / (SEQL + 4));
    int l = lp - 2;
    float v = (l >= 0 && l < SEQL)
                  ? x[((size_t)bch * SEQL + l) * DMOD + dcol] : 0.f;
    xpad[idx] = (__bf16)v;
}

// ---------------------------------------------------------------------------
// Host launcher
// ---------------------------------------------------------------------------
extern "C" void kernel_launch(void* const* d_in, const int* in_sizes, int n_in,
                              void* d_out, int out_size, void* d_ws, size_t ws_size,
                              hipStream_t stream)
{
    const float* x         = (const float*)d_in[0];
    const float* ln1_w     = (const float*)d_in[1];
    const float* ln1_b     = (const float*)d_in[2];
    const float* in_proj_w = (const float*)d_in[3];
    const float* conv_w    = (const float*)d_in[4];
    const float* conv_b    = (const float*)d_in[5];
    const float* x_proj_w  = (const float*)d_in[6];
    const float* dt_proj_w = (const float*)d_in[7];
    const float* dt_proj_b = (const float*)d_in[8];
    const float* A_log     = (const float*)d_in[9];
    const float* D_param   = (const float*)d_in[10];
    const float* out_proj_w= (const float*)d_in[11];
    const float* ln2_w     = (const float*)d_in[12];
    const float* ln2_b     = (const float*)d_in[13];
    const float* convL_w   = (const float*)d_in[14];
    const float* convL_b   = (const float*)d_in[15];
    const float* mlp_w1    = (const float*)d_in[16];
    const float* mlp_b1    = (const float*)d_in[17];
    const float* mlp_w2    = (const float*)d_in[18];
    const float* mlp_b2    = (const float*)d_in[19];
    float* out = (float*)d_out;

    char* base = (char*)d_ws;
    size_t off = 0;
    auto alloc = [&](size_t bytes) -> char* {
        char* p = base + off;
        off += (bytes + 255) & ~(size_t)255;
        return p;
    };

    __bf16* u_bf     = (__bf16*)alloc((size_t)ROWS * DMOD * 2);
    __bf16* xz_bf    = (__bf16*)alloc((size_t)ROWS * 2 * DIN * 2);
    __bf16* xc_bf    = (__bf16*)alloc((size_t)ROWS * DIN * 2);
    float*  xdbl_f   = (float*) alloc((size_t)ROWS * 96 * 4);
    __bf16* xdbl_bf  = (__bf16*)alloc((size_t)ROWS * 96 * 2);
    float*  dt_f     = (float*) alloc((size_t)ROWS * DIN * 4);
    __bf16* y_bf     = (__bf16*)alloc((size_t)ROWS * DIN * 2);
    __bf16* xpad_bf  = (__bf16*)alloc((size_t)BATCH * (SEQL + 4) * DMOD * 2);
    float*  local_f  = (float*) alloc((size_t)ROWS * DMOD * 4);
    float*  h_f      = (float*) alloc((size_t)ROWS * DMOD * 4);
    __bf16* ln2_bf   = (__bf16*)alloc((size_t)ROWS * DMOD * 2);
    __bf16* m1_bf    = (__bf16*)alloc((size_t)ROWS * HMLP * 2);

    __bf16* w_inproj = (__bf16*)alloc((size_t)2 * DIN * DMOD * 2);
    __bf16* w_xproj  = (__bf16*)alloc((size_t)96 * DIN * 2);
    __bf16* w_dtproj = (__bf16*)alloc((size_t)DIN * RRANK * 2);
    __bf16* w_outproj= (__bf16*)alloc((size_t)DMOD * DIN * 2);
    __bf16* w_convL  = (__bf16*)alloc((size_t)5 * DMOD * DMOD * 2);
    __bf16* w_mlp1   = (__bf16*)alloc((size_t)HMLP * DMOD * 2);
    __bf16* w_mlp2   = (__bf16*)alloc((size_t)DMOD * HMLP * 2);

    const int NOREMAP = 1 << 30;
    auto cvt = [&](const float* src, __bf16* dst, int n) {
        cvt_f32_bf16<<<dim3((n + 255) / 256), dim3(256), 0, stream>>>(src, dst, n);
    };

    // --- weight conversions (bf16 for WMMA) ---
    cvt(in_proj_w, w_inproj, 2 * DIN * DMOD);
    cvt(x_proj_w,  w_xproj,  96 * DIN);
    cvt(dt_proj_w, w_dtproj, DIN * RRANK);
    cvt(out_proj_w,w_outproj,DMOD * DIN);
    cvt(mlp_w1,    w_mlp1,   HMLP * DMOD);
    cvt(mlp_w2,    w_mlp2,   DMOD * HMLP);
    cvt_convL<<<dim3((5 * DMOD * DMOD + 255) / 256), dim3(256), 0, stream>>>(convL_w, w_convL);
    pad_x_bf<<<dim3((BATCH * (SEQL + 4) * DMOD + 255) / 256), dim3(256), 0, stream>>>(x, xpad_bf);

    // --- LN1 ---
    layernorm_bf<<<dim3(ROWS), dim3(256), 0, stream>>>(x, ln1_w, ln1_b, u_bf);

    // --- in_proj: (4096,1024) x (4096,1024)^T -> xz (bf16 only) ---
    gemm_wmma_bf16<<<dim3((2 * DIN) / 128, ROWS / 128), dim3(256), 0, stream>>>(
        u_bf, w_inproj, nullptr, xz_bf, nullptr, nullptr,
        ROWS, 2 * DIN, DMOD, DMOD, DMOD, 2 * DIN,
        NOREMAP, 0, 0, 0, 0);

    // --- depthwise conv + SiLU ---
    dwconv_silu<<<dim3((ROWS * DIN) / 256), dim3(256), 0, stream>>>(
        xz_bf, conv_w, conv_b, xc_bf);

    // --- x_proj: (4096,2048) -> (4096,96) f32 + bf16 ---
    gemm_wmma_bf16<<<dim3(1, ROWS / 128), dim3(256), 0, stream>>>(
        xc_bf, w_xproj, xdbl_f, xdbl_bf, nullptr, nullptr,
        ROWS, 96, DIN, DIN, DIN, 96,
        NOREMAP, 0, 0, 0, 0);

    // --- dt_proj + softplus: (4096,64) -> (4096,2048) f32 ---
    gemm_wmma_bf16<<<dim3(DIN / 128, ROWS / 128), dim3(256), 0, stream>>>(
        xdbl_bf, w_dtproj, dt_f, nullptr, dt_proj_b, nullptr,
        ROWS, DIN, RRANK, 96, RRANK, DIN,
        NOREMAP, 0, 0, 1, 0);

    // --- selective scan (sequential in L, parallel over B x DIN) ---
    selective_scan<<<dim3(DIN / 256, BATCH), dim3(256), 0, stream>>>(
        dt_f, xdbl_f, xc_bf, xz_bf, A_log, D_param, y_bf);

    // --- local dense conv1d as 5 accumulated GEMMs over padded x ---
    for (int t = 0; t < 5; ++t) {
        gemm_wmma_bf16<<<dim3(DMOD / 128, ROWS / 128), dim3(256), 0, stream>>>(
            xpad_bf, w_convL + (size_t)t * DMOD * DMOD,
            local_f, nullptr, (t == 0) ? convL_b : nullptr, nullptr,
            ROWS, DMOD, DMOD, DMOD, DMOD, DMOD,
            SEQL, 4, t, 0, (t == 0) ? 0 : 1);
    }

    // --- out_proj + residual local -> h ---
    gemm_wmma_bf16<<<dim3(DMOD / 128, ROWS / 128), dim3(256), 0, stream>>>(
        y_bf, w_outproj, h_f, nullptr, nullptr, local_f,
        ROWS, DMOD, DIN, DIN, DIN, DMOD,
        NOREMAP, 0, 0, 0, 0);

    // --- LN2 ---
    layernorm_bf<<<dim3(ROWS), dim3(256), 0, stream>>>(h_f, ln2_w, ln2_b, ln2_bf);

    // --- MLP up + GELU (bf16 only output) ---
    gemm_wmma_bf16<<<dim3(HMLP / 128, ROWS / 128), dim3(256), 0, stream>>>(
        ln2_bf, w_mlp1, nullptr, m1_bf, mlp_b1, nullptr,
        ROWS, HMLP, DMOD, DMOD, DMOD, HMLP,
        NOREMAP, 0, 0, 2, 0);

    // --- MLP down + bias + residual h -> d_out ---
    gemm_wmma_bf16<<<dim3(DMOD / 128, ROWS / 128), dim3(256), 0, stream>>>(
        m1_bf, w_mlp2, out, nullptr, mlp_b2, h_f,
        ROWS, DMOD, HMLP, HMLP, HMLP, DMOD,
        NOREMAP, 0, 0, 0, 0);
}
